// ImprovedDiscriminator_16758962389451
// MI455X (gfx1250) — matrix operand order repaired and checked by reading, not compiled, not run on hardware
//
#include <hip/hip_runtime.h>
#include <hip/hip_bf16.h>

typedef __attribute__((ext_vector_type(16))) _Float16 v16h;
typedef __attribute__((ext_vector_type(8)))  _Float16 v8h;
typedef __attribute__((ext_vector_type(8)))  float    v8f;

__device__ __forceinline__ float lrelu(float v) { return v > 0.0f ? v : 0.2f * v; }

// ---------------------------------------------------------------------------
// Small packing / conversion kernels
// ---------------------------------------------------------------------------
__global__ void k_cvt_f32_to_f16(const float* __restrict__ src,
                                 _Float16* __restrict__ dst, int n) {
  for (int i = blockIdx.x * blockDim.x + threadIdx.x; i < n;
       i += gridDim.x * blockDim.x)
    dst[i] = (_Float16)src[i];
}

// conv1 weights [64, 3*4*4=48] -> f16 [64, 64] with K zero-padded to 64
__global__ void k_wpack1(const float* __restrict__ w, _Float16* __restrict__ A1) {
  int i = blockIdx.x * blockDim.x + threadIdx.x;
  if (i >= 64 * 64) return;
  int m = i >> 6, k = i & 63;
  A1[i] = (k < 48) ? (_Float16)w[m * 48 + k] : (_Float16)0.0f;
}

// T [4096, 16, 16] -> Tt f16 [256, 4096]  (Tt[of][k] = T[k][of])
__global__ void k_tpack(const float* __restrict__ T, _Float16* __restrict__ Tt) {
  for (long e = blockIdx.x * blockDim.x + threadIdx.x; e < 256L * 4096;
       e += (long)gridDim.x * blockDim.x) {
    int of = (int)(e >> 12);
    int k  = (int)(e & 4095);
    Tt[e] = (_Float16)T[(long)k * 256 + of];
  }
}

// ---------------------------------------------------------------------------
// im2col kernels (produce Bt layout: col[n*K + k], f16)
// ---------------------------------------------------------------------------
// conv1: img NCHW [512,3,32,32] -> col [131072, 64]  (n = b*256 + oh*16 + ow)
__global__ void k_im2col1(const float* __restrict__ img, _Float16* __restrict__ col) {
  const long total = 131072L * 64;
  for (long e = blockIdx.x * blockDim.x + threadIdx.x; e < total;
       e += (long)gridDim.x * blockDim.x) {
    int k = (int)(e & 63); long n = e >> 6;
    _Float16 v = (_Float16)0.0f;
    if (k < 48) {
      int c = k >> 4, kh = (k >> 2) & 3, kw = k & 3;
      int b = (int)(n >> 8), hw = (int)(n & 255), oh = hw >> 4, ow = hw & 15;
      int ih = oh * 2 - 1 + kh, iw = ow * 2 - 1 + kw;
      if (ih >= 0 && ih < 32 && iw >= 0 && iw < 32)
        v = (_Float16)img[(((long)b * 3 + c) * 32 + ih) * 32 + iw];
    }
    col[e] = v;
  }
}

// conv2: x1 f16 [c=64][b=512][16][16] -> col [32768, 1024] (n = b*64 + oh*8 + ow)
__global__ void k_im2col2(const _Float16* __restrict__ x1, _Float16* __restrict__ col) {
  const long total = 32768L * 1024;
  for (long e = blockIdx.x * blockDim.x + threadIdx.x; e < total;
       e += (long)gridDim.x * blockDim.x) {
    int k = (int)(e & 1023); long n = e >> 10;
    int c = k >> 4, kh = (k >> 2) & 3, kw = k & 3;
    int b = (int)(n >> 6), hw = (int)(n & 63), oh = hw >> 3, ow = hw & 7;
    int ih = oh * 2 - 1 + kh, iw = ow * 2 - 1 + kw;
    _Float16 v = (_Float16)0.0f;
    if (ih >= 0 && ih < 16 && iw >= 0 && iw < 16)
      v = x1[(((long)c * 512 + b) * 16 + ih) * 16 + iw];
    col[e] = v;
  }
}

// conv3: raw y2 f32 [c=128][b*64 + ih*8 + iw]; apply BN(scale,shift)+lrelu on the fly
// -> col [8192, 2048]  (n = b*16 + oh*4 + ow)
__global__ void k_im2col3(const float* __restrict__ y2, const float* __restrict__ bnp,
                          _Float16* __restrict__ col) {
  const long total = 8192L * 2048;
  for (long e = blockIdx.x * blockDim.x + threadIdx.x; e < total;
       e += (long)gridDim.x * blockDim.x) {
    int k = (int)(e & 2047); long n = e >> 11;
    int c = k >> 4, kh = (k >> 2) & 3, kw = k & 3;
    int b = (int)(n >> 4), hw = (int)(n & 15), oh = hw >> 2, ow = hw & 3;
    int ih = oh * 2 - 1 + kh, iw = ow * 2 - 1 + kw;
    _Float16 v = (_Float16)0.0f;
    if (ih >= 0 && ih < 8 && iw >= 0 && iw < 8) {
      float raw = y2[(long)c * 32768 + b * 64 + ih * 8 + iw];
      v = (_Float16)lrelu(bnp[c * 2] * raw + bnp[c * 2 + 1]);
    }
    col[e] = v;
  }
}

// ---------------------------------------------------------------------------
// BatchNorm statistics: one block per channel, f32, biased variance.
// Outputs per channel: bnp[2c] = rstd*gamma, bnp[2c+1] = beta - mean*rstd*gamma
// ---------------------------------------------------------------------------
__global__ void k_bnstats(const float* __restrict__ y, const float* __restrict__ gamma,
                          const float* __restrict__ beta, float* __restrict__ bnp,
                          int npix) {
  int c = blockIdx.x;
  const float* p = y + (long)c * npix;
  float s = 0.0f, s2 = 0.0f;
  for (int i = threadIdx.x; i < npix; i += blockDim.x) {
    float v = p[i]; s += v; s2 += v * v;
  }
  __shared__ float sh[512];
  sh[threadIdx.x] = s; sh[256 + threadIdx.x] = s2;
  __syncthreads();
  for (int off = 128; off > 0; off >>= 1) {
    if ((int)threadIdx.x < off) {
      sh[threadIdx.x] += sh[threadIdx.x + off];
      sh[256 + threadIdx.x] += sh[256 + threadIdx.x + off];
    }
    __syncthreads();
  }
  if (threadIdx.x == 0) {
    float inv = 1.0f / (float)npix;
    float mean = sh[0] * inv;
    float var  = sh[256] * inv - mean * mean;
    float rstd = rsqrtf(var + 1e-5f);
    float scale = rstd * gamma[c];
    bnp[c * 2] = scale;
    bnp[c * 2 + 1] = beta[c] - mean * scale;
  }
}

// BN-apply + lrelu on y3 and pack feats [b][c*16+hw] as both f32 and f16
__global__ void k_bnapply3(const float* __restrict__ y3, const float* __restrict__ bnp,
                           float* __restrict__ f32o, _Float16* __restrict__ f16o) {
  const long total = 512L * 4096;
  for (long e = blockIdx.x * blockDim.x + threadIdx.x; e < total;
       e += (long)gridDim.x * blockDim.x) {
    int b = (int)(e >> 12); int f = (int)(e & 4095);
    int c = f >> 4; int hw = f & 15;
    float v = y3[(long)c * 8192 + b * 16 + hw];
    v = lrelu(bnp[c * 2] * v + bnp[c * 2 + 1]);
    f32o[e] = v;
    f16o[e] = (_Float16)v;
  }
}

// ---------------------------------------------------------------------------
// WMMA GEMM: D[M,N] f32 = A[M,K] f16 (row-major) x Bt[N,K] f16 (B transposed).
// One wave per 16x64 output tile (4 accumulators, A-fragment reuse).
// Fragment layouts used by this code (wave32, contiguous k order, same for A and B):
//   A 16x32 f16: elem j    -> K=16g+j;                            row = lane&15
//   B 32x16 f16: elem j    -> K=16g+j;                            col = lane&15
//   C/D 16x16 f32: vgpr i -> row = i + 8g; col = lane&15
// EPI=0: store raw f32.  EPI=1: lrelu + store f16.
// ---------------------------------------------------------------------------
template <int EPI>
__global__ __launch_bounds__(256)
void k_gemm(const _Float16* __restrict__ A, const _Float16* __restrict__ Bt,
            float* __restrict__ Df32, _Float16* __restrict__ Df16,
            int M, int N, int K) {
  const int lane = threadIdx.x & 31;
  const int wave = blockIdx.x * (blockDim.x >> 5) + (threadIdx.x >> 5);
  const int ntiles = N >> 6;
  if (wave >= (M >> 4) * ntiles) return;
  const int m0 = (wave / ntiles) << 4;
  const int n0 = (wave % ntiles) << 6;
  const int g = lane >> 4;
  const int r = lane & 15;

  const _Float16* ap  = A  + (size_t)(m0 + r) * K + 16 * g;
  const _Float16* bp0 = Bt + (size_t)(n0 +      r) * K + 16 * g;
  const _Float16* bp1 = Bt + (size_t)(n0 + 16 + r) * K + 16 * g;
  const _Float16* bp2 = Bt + (size_t)(n0 + 32 + r) * K + 16 * g;
  const _Float16* bp3 = Bt + (size_t)(n0 + 48 + r) * K + 16 * g;

  v8f c0 = {}, c1 = {}, c2 = {}, c3 = {};
  for (int k0 = 0; k0 < K; k0 += 32) {
    v8h alo = *(const v8h*)(ap + k0);
    v8h ahi = *(const v8h*)(ap + k0 + 8);
    v16h a = __builtin_shufflevector(alo, ahi, 0, 1, 2, 3, 4, 5, 6, 7,
                                     8, 9, 10, 11, 12, 13, 14, 15);
    v16h b0 = *(const v16h*)(bp0 + k0);
    v16h b1 = *(const v16h*)(bp1 + k0);
    v16h b2 = *(const v16h*)(bp2 + k0);
    v16h b3 = *(const v16h*)(bp3 + k0);
    c0 = __builtin_amdgcn_wmma_f32_16x16x32_f16(false, a, false, b0, (short)0, c0, false, false);
    c1 = __builtin_amdgcn_wmma_f32_16x16x32_f16(false, a, false, b1, (short)0, c1, false, false);
    c2 = __builtin_amdgcn_wmma_f32_16x16x32_f16(false, a, false, b2, (short)0, c2, false, false);
    c3 = __builtin_amdgcn_wmma_f32_16x16x32_f16(false, a, false, b3, (short)0, c3, false, false);
  }

#pragma unroll
  for (int i = 0; i < 8; i++) {
    int m = m0 + i + 8 * g;
    size_t base = (size_t)m * N + n0 + r;
    if (EPI == 0) {
      Df32[base]      = c0[i];
      Df32[base + 16] = c1[i];
      Df32[base + 32] = c2[i];
      Df32[base + 48] = c3[i];
    } else {
      Df16[base]      = (_Float16)lrelu(c0[i]);
      Df16[base + 16] = (_Float16)lrelu(c1[i]);
      Df16[base + 32] = (_Float16)lrelu(c2[i]);
      Df16[base + 48] = (_Float16)lrelu(c3[i]);
    }
  }
}

// ---------------------------------------------------------------------------
// Minibatch discrimination: o[b,o] = sum_{b2 != b} exp(-sum_i |M[b,o,i]-M[b2,o,i]|)
// One block per row b; thread = (group, o); M row cached in LDS.
// ---------------------------------------------------------------------------
__global__ void k_mbdisc(const float* __restrict__ Mm, float* __restrict__ oo) {
  int b = blockIdx.x;
  __shared__ float sM[256];
  __shared__ float sP[256];
  sM[threadIdx.x] = Mm[(long)b * 256 + threadIdx.x];
  __syncthreads();
  int o = threadIdx.x & 15;
  int grp = threadIdx.x >> 4;
  float acc = 0.0f;
  for (int b2 = grp; b2 < 512; b2 += 16) {
    if (b2 == b) continue;
    const float* q = Mm + (long)b2 * 256 + o * 16;
    float d = 0.0f;
#pragma unroll
    for (int i = 0; i < 16; i++) d += fabsf(sM[o * 16 + i] - q[i]);
    acc += __expf(-d);
  }
  sP[threadIdx.x] = acc;
  __syncthreads();
  if (threadIdx.x < 16) {
    float s = 0.0f;
#pragma unroll
    for (int gg = 0; gg < 16; gg++) s += sP[gg * 16 + threadIdx.x];
    oo[b * 16 + threadIdx.x] = s;
  }
}

// Final linear (4112 -> 1) + sigmoid; one block per batch element
__global__ void k_final(const float* __restrict__ feats, const float* __restrict__ oo,
                        const float* __restrict__ w, const float* __restrict__ bias,
                        float* __restrict__ out) {
  int b = blockIdx.x;
  float acc = 0.0f;
  for (int k = threadIdx.x; k < 4112; k += 256) {
    float x = (k < 4096) ? feats[(long)b * 4096 + k] : oo[b * 16 + (k - 4096)];
    acc += x * w[k];
  }
  __shared__ float sh[256];
  sh[threadIdx.x] = acc;
  __syncthreads();
  for (int off = 128; off > 0; off >>= 1) {
    if ((int)threadIdx.x < off) sh[threadIdx.x] += sh[threadIdx.x + off];
    __syncthreads();
  }
  if (threadIdx.x == 0) {
    float z = sh[0] + bias[0];
    out[b] = 1.0f / (1.0f + __expf(-z));
  }
}

// ---------------------------------------------------------------------------
// Launcher
// ---------------------------------------------------------------------------
extern "C" void kernel_launch(void* const* d_in, const int* in_sizes, int n_in,
                              void* d_out, int out_size, void* d_ws, size_t ws_size,
                              hipStream_t stream) {
  (void)in_sizes; (void)n_in; (void)out_size; (void)ws_size;
  const float* img = (const float*)d_in[0];
  const float* w1  = (const float*)d_in[1];
  const float* w2  = (const float*)d_in[2];
  const float* w3  = (const float*)d_in[3];
  const float* g2  = (const float*)d_in[4];
  const float* b2  = (const float*)d_in[5];
  const float* g3  = (const float*)d_in[6];
  const float* b3  = (const float*)d_in[7];
  const float* T   = (const float*)d_in[8];
  const float* lw  = (const float*)d_in[9];
  const float* lb  = (const float*)d_in[10];
  float* out = (float*)d_out;

  char* ws = (char*)d_ws;
  const size_t KB = 1024ull, MB = 1024ull * 1024ull;
  _Float16* A1   = (_Float16*)(ws + 0);            //   8 KB
  _Float16* A2   = (_Float16*)(ws + 16 * KB);      // 256 KB
  _Float16* A3   = (_Float16*)(ws + 288 * KB);     //   1 MB
  _Float16* Tt   = (_Float16*)(ws + 1344 * KB);    //   2 MB
  float*    bnp2 = (float*)   (ws + 3392 * KB);    //   1 KB
  float*    bnp3 = (float*)   (ws + 3396 * KB);    //   2 KB
  float*    oout = (float*)   (ws + 3400 * KB);    //  32 KB
  float*    Mmat = (float*)   (ws + 3436 * KB);    // 512 KB
  const size_t BIG = 4 * MB;
  _Float16* col1 = (_Float16*)(ws + BIG);           // 16 MB
  _Float16* x1   = (_Float16*)(ws + BIG + 16 * MB); // 16 MB
  _Float16* col2 = (_Float16*)(ws + BIG + 32 * MB); // 64 MB
  float*    y2   = (float*)   (ws + BIG + 96 * MB); // 16 MB
  _Float16* col3 = (_Float16*)(ws + BIG);           // 32 MB (reuses col1+x1)
  float*    y3   = (float*)   (ws + BIG + 32 * MB); //  8 MB (reuses col2)
  float*    ff32 = (float*)   (ws + BIG + 40 * MB); //  8 MB
  _Float16* ff16 = (_Float16*)(ws + BIG + 48 * MB); //  4 MB

  // --- weight packing ---
  k_wpack1<<<16, 256, 0, stream>>>(w1, A1);
  k_cvt_f32_to_f16<<<512, 256, 0, stream>>>(w2, A2, 128 * 1024);
  k_cvt_f32_to_f16<<<2048, 256, 0, stream>>>(w3, A3, 256 * 2048);
  k_tpack<<<2048, 256, 0, stream>>>(T, Tt);

  // --- conv1 (GEMM 64 x 131072 x 64) + lrelu -> x1 f16 [64][131072] ---
  k_im2col1<<<8192, 256, 0, stream>>>(img, col1);
  k_gemm<1><<<1024, 256, 0, stream>>>(A1, col1, (float*)nullptr, x1, 64, 131072, 64);

  // --- conv2 (GEMM 128 x 32768 x 1024) -> raw y2, BN stats ---
  k_im2col2<<<8192, 256, 0, stream>>>(x1, col2);
  k_gemm<0><<<512, 256, 0, stream>>>(A2, col2, y2, (_Float16*)nullptr, 128, 32768, 1024);
  k_bnstats<<<128, 256, 0, stream>>>(y2, g2, b2, bnp2, 32768);

  // --- conv3 (BN2+lrelu fused into im2col; GEMM 256 x 8192 x 2048) ---
  k_im2col3<<<8192, 256, 0, stream>>>(y2, bnp2, col3);
  k_gemm<0><<<256, 256, 0, stream>>>(A3, col3, y3, (_Float16*)nullptr, 256, 8192, 2048);
  k_bnstats<<<256, 256, 0, stream>>>(y3, g3, b3, bnp3, 8192);
  k_bnapply3<<<4096, 256, 0, stream>>>(y3, bnp3, ff32, ff16);

  // --- minibatch discrimination: M = feats @ T  (GEMM 512 x 256 x 4096) ---
  k_gemm<0><<<16, 256, 0, stream>>>(ff16, Tt, Mmat, (_Float16*)nullptr, 512, 256, 4096);
  k_mbdisc<<<512, 256, 0, stream>>>(Mmat, oout);

  // --- final linear + sigmoid ---
  k_final<<<512, 256, 0, stream>>>(ff32, oout, lw, lb, out);
}
